// SDARAttention_10514079941147
// MI455X (gfx1250) — matrix operand order repaired and checked by reading, not compiled, not run on hardware
//
#include <hip/hip_runtime.h>
#include <hip/hip_bf16.h>

// ---------------------------------------------------------------------------
// Problem constants (match reference)
// ---------------------------------------------------------------------------
#define BB   2
#define SS   2048
#define HID  2048
#define NH   16
#define NKV  4
#define DD   128
#define NQKV 3072            // (NH + 2*NKV) * DD
#define MTOT 4096            // BB * SS

typedef _Float16 v16h __attribute__((ext_vector_type(16)));
typedef _Float16 h8v  __attribute__((ext_vector_type(8)));
typedef _Float16 h4v  __attribute__((ext_vector_type(4)));
typedef float    v8f  __attribute__((ext_vector_type(8)));

// ---------------------------------------------------------------------------
// WMMA helpers (CDNA5 wave32; V_WMMA_F32_16X16X32_F16)
// ---------------------------------------------------------------------------
__device__ __forceinline__ v8f wmma16(v16h a, v16h b, v8f c) {
    return __builtin_amdgcn_wmma_f32_16x16x32_f16(
        false, a, false, b, (short)0, c, false, false);
}

// A-matrix 16x32 fragment (16-bit): lane<16 holds K {0..7, 16..23},
// lane>=16 holds K {8..15, 24..31} of row (lane&15).  rowp = &A[row][0].
__device__ __forceinline__ v16h a_frag(const _Float16* __restrict__ rowp,
                                       int k, int lane) {
    const int kb = k + ((lane >> 4) << 3);
    h8v lo = *reinterpret_cast<const h8v*>(rowp + kb);
    h8v hi = *reinterpret_cast<const h8v*>(rowp + kb + 16);
    v16h r;
#pragma unroll
    for (int i = 0; i < 8; ++i) { r[i] = lo[i]; r[i + 8] = hi[i]; }
    return r;
}

// B-matrix 32x16 fragment (16-bit): lane = column (lane&15); lane<16 holds
// K {0..7, 16..23}, lane>=16 holds K {8..15, 24..31}.  colp = &B[:, col] base along K.
__device__ __forceinline__ v16h b_frag(const _Float16* __restrict__ colp,
                                       int k, int lane) {
    const int kb = k + ((lane >> 4) << 3);
    h8v lo = *reinterpret_cast<const h8v*>(colp + kb);
    h8v hi = *reinterpret_cast<const h8v*>(colp + kb + 16);
    v16h r;
#pragma unroll
    for (int i = 0; i < 8; ++i) { r[i] = lo[i]; r[i + 8] = hi[i]; }
    return r;
}

// ---------------------------------------------------------------------------
// Kernel 0: f32 -> f16 conversion (vectorized 4-wide)
// ---------------------------------------------------------------------------
__global__ void f32_to_f16_kernel(const float* __restrict__ in,
                                  _Float16* __restrict__ out, int n4) {
    int i = blockIdx.x * 256 + threadIdx.x;
    if (i < n4) {
        float4 v = reinterpret_cast<const float4*>(in)[i];
        h4v h;
        h[0] = (_Float16)v.x; h[1] = (_Float16)v.y;
        h[2] = (_Float16)v.z; h[3] = (_Float16)v.w;
        reinterpret_cast<h4v*>(out)[i] = h;
    }
}

// ---------------------------------------------------------------------------
// Kernel 1: NT GEMM  C[M,N](f32) = A[M,K](f16, row-major) * Bw[N,K](f16)^T
// Block: 256 threads = 8 waves (2 x 4); wave tile 32x64; block tile 64x256.
// Per K-step: 2 A-frags + 4 B-frags (192 B/lane) feed 8 WMMAs; each A frag
// is reused 4x and each B frag 2x, maximizing matrix-unit occupancy per byte.
// ---------------------------------------------------------------------------
__global__ __launch_bounds__(256)
void gemm_f16_nt(const _Float16* __restrict__ A,
                 const _Float16* __restrict__ Bw,
                 float* __restrict__ C, int M, int N, int K) {
    const int lane = threadIdx.x & 31;
    const int wv   = threadIdx.x >> 5;
    const int m0   = blockIdx.y * 64  + (wv >> 2) * 32;
    const int n0   = blockIdx.x * 256 + (wv & 3) * 64;
    const int mr   = lane & 15;

    const _Float16* a0p = A  + (size_t)(m0 + mr)      * K;
    const _Float16* a1p = A  + (size_t)(m0 + 16 + mr) * K;
    const _Float16* bp[4];
#pragma unroll
    for (int j = 0; j < 4; ++j)
        bp[j] = Bw + (size_t)(n0 + j * 16 + mr) * K;

    v8f acc[2][4];
#pragma unroll
    for (int i = 0; i < 2; ++i)
#pragma unroll
        for (int j = 0; j < 4; ++j) acc[i][j] = (v8f){};

#pragma unroll 2
    for (int k = 0; k < K; k += 32) {
        v16h A0 = a_frag(a0p, k, lane);
        v16h A1 = a_frag(a1p, k, lane);
        v16h Bf[4];
#pragma unroll
        for (int j = 0; j < 4; ++j) Bf[j] = b_frag(bp[j], k, lane);
#pragma unroll
        for (int j = 0; j < 4; ++j) {
            acc[0][j] = wmma16(A0, Bf[j], acc[0][j]);
            acc[1][j] = wmma16(A1, Bf[j], acc[1][j]);
        }
    }
    // C/D layout: VGPR r -> row r (lanes 0-15) / row r+8 (lanes 16-31), col = lane&15
    const int rowo = (lane >> 4) << 3;
    const int col  = lane & 15;
#pragma unroll
    for (int i = 0; i < 2; ++i)
#pragma unroll
        for (int r = 0; r < 8; ++r) {
            size_t rb = (size_t)(m0 + i * 16 + rowo + r) * N + n0 + col;
#pragma unroll
            for (int j = 0; j < 4; ++j)
                C[rb + j * 16] = acc[i][j][r];
        }
}

// ---------------------------------------------------------------------------
// Kernel 2: per-head RMSNorm + RoPE + pack to f16.  One wave per (token, head).
// Heads 0..15 = Q (scaled by D^-1/2), 16..19 = K, 20..23 = V (V stored
// transposed [b, kv, d, s] so attention PV B-fragments are contiguous).
// ---------------------------------------------------------------------------
__global__ __launch_bounds__(256)
void pack_qkv_kernel(const float* __restrict__ qkv,
                     const int* __restrict__ pos,
                     const float* __restrict__ qw,
                     const float* __restrict__ kw,
                     _Float16* __restrict__ qo,
                     _Float16* __restrict__ ko,
                     _Float16* __restrict__ vT) {
    const int lane = threadIdx.x & 31;
    const int g    = blockIdx.x * 8 + (threadIdx.x >> 5);
    const int m    = g / 24;            // token (b*S + s)
    const int h    = g - m * 24;        // head slot 0..23
    const float* base = qkv + (size_t)m * NQKV + h * DD;

    const int d0 = lane * 2;            // 0..62
    float x0 = base[d0], x1 = base[d0 + 1];
    float x2 = base[64 + d0], x3 = base[64 + d0 + 1];

    float ss = x0 * x0 + x1 * x1 + x2 * x2 + x3 * x3;
#pragma unroll
    for (int off = 16; off >= 1; off >>= 1) ss += __shfl_xor(ss, off, 32);

    const int b = m >> 11;              // / SS
    const int s = m & (SS - 1);

    if (h < NH + NKV) {                 // Q or K: norm + rope
        float rn = rsqrtf(ss * (1.0f / 128.0f) + 1e-6f);
        const float* w = (h < NH) ? qw : kw;
        x0 *= rn * w[d0];       x1 *= rn * w[d0 + 1];
        x2 *= rn * w[64 + d0];  x3 *= rn * w[64 + d0 + 1];

        float p = (float)pos[m];
        const float LN = 9.2103403719761836f;   // ln(10000)
        float a0 = p * __expf(-LN * (float)d0 * (1.0f / 64.0f));
        float a1 = p * __expf(-LN * (float)(d0 + 1) * (1.0f / 64.0f));
        float c0 = __cosf(a0), s0 = __sinf(a0);
        float c1 = __cosf(a1), s1 = __sinf(a1);
        float o0 = x0 * c0 - x2 * s0;
        float o1 = x1 * c1 - x3 * s1;
        float o2 = x2 * c0 + x0 * s0;
        float o3 = x3 * c1 + x1 * s1;

        if (h < NH) {
            const float sc = 0.08838834764831845f;  // 128^-0.5 folded into Q
            o0 *= sc; o1 *= sc; o2 *= sc; o3 *= sc;
            _Float16* q = qo + ((size_t)m * NH + h) * DD;
            q[d0] = (_Float16)o0;      q[d0 + 1] = (_Float16)o1;
            q[64 + d0] = (_Float16)o2; q[65 + d0] = (_Float16)o3;
        } else {
            _Float16* kk = ko + ((size_t)m * NKV + (h - NH)) * DD;
            kk[d0] = (_Float16)o0;      kk[d0 + 1] = (_Float16)o1;
            kk[64 + d0] = (_Float16)o2; kk[65 + d0] = (_Float16)o3;
        }
    } else {                            // V: passthrough, transposed store
        const int kh = h - (NH + NKV);
        _Float16* vb = vT + (size_t)(b * NKV + kh) * DD * SS;
        vb[(size_t)d0 * SS + s]        = (_Float16)x0;
        vb[(size_t)(d0 + 1) * SS + s]  = (_Float16)x1;
        vb[(size_t)(64 + d0) * SS + s] = (_Float16)x2;
        vb[(size_t)(65 + d0) * SS + s] = (_Float16)x3;
    }
}

// ---------------------------------------------------------------------------
// Kernel 3: flash attention (non-causal, GQA).  grid = (S/128, NH, B),
// block = 256 (8 waves); wave owns a 16x128 Q tile, streams K/V in 32-key
// steps: 8 QK^T WMMAs + online softmax + LDS transpose of P + 8 PV WMMAs.
// ---------------------------------------------------------------------------
__global__ __launch_bounds__(256)
void attn_kernel(const _Float16* __restrict__ q,
                 const _Float16* __restrict__ k,
                 const _Float16* __restrict__ vT,
                 _Float16* __restrict__ ctx) {
    __shared__ float pbuf[8][16 * 33];  // per-wave P transpose buffer (padded)

    const int lane = threadIdx.x & 31;
    const int wv   = threadIdx.x >> 5;
    const int b    = blockIdx.z;
    const int h    = blockIdx.y;
    const int kh   = h >> 2;            // GQA: NH/NKV = 4
    const int q0   = blockIdx.x * 128 + wv * 16;
    const int mr   = lane & 15;
    const int half = lane >> 4;

    // Q tile -> 4 A-fragments (K = d, 128 total)
    const _Float16* qrow = q + ((size_t)(b * SS + q0 + mr) * NH + h) * DD;
    v16h qa[4];
#pragma unroll
    for (int kk = 0; kk < 4; ++kk) qa[kk] = a_frag(qrow, kk * 32, lane);

    v8f o[8] = {};                      // 16x128 fp32 accumulator
    float mstat[8], lstat[8];
#pragma unroll
    for (int r = 0; r < 8; ++r) { mstat[r] = -1e30f; lstat[r] = 0.0f; }

    float* pb = pbuf[wv];

    for (int j0 = 0; j0 < SS; j0 += 32) {
        // ---- scores: S[16q x 32kv] = Q (16x128) * K^T
        v8f s0 = {}, s1 = {};
        const _Float16* k0p = k + ((size_t)(b * SS + j0 + mr)      * NKV + kh) * DD;
        const _Float16* k1p = k + ((size_t)(b * SS + j0 + 16 + mr) * NKV + kh) * DD;
#pragma unroll
        for (int kk = 0; kk < 4; ++kk) {
            v16h kb0 = b_frag(k0p, kk * 32, lane);
            v16h kb1 = b_frag(k1p, kk * 32, lane);
            s0 = wmma16(qa[kk], kb0, s0);
            s1 = wmma16(qa[kk], kb1, s1);
        }

        // ---- online softmax (row stats per 16-lane half)
        float al[8];
#pragma unroll
        for (int r = 0; r < 8; ++r) {
            float mx = fmaxf(s0[r], s1[r]);
#pragma unroll
            for (int off = 8; off >= 1; off >>= 1)
                mx = fmaxf(mx, __shfl_xor(mx, off, 32));
            float mn = fmaxf(mstat[r], mx);
            al[r] = __expf(mstat[r] - mn);
            float p0 = __expf(s0[r] - mn);
            float p1 = __expf(s1[r] - mn);
            s0[r] = p0; s1[r] = p1;
            float rs = p0 + p1;
#pragma unroll
            for (int off = 8; off >= 1; off >>= 1)
                rs += __shfl_xor(rs, off, 32);
            lstat[r] = lstat[r] * al[r] + rs;
            mstat[r] = mn;
        }
#pragma unroll
        for (int dd = 0; dd < 8; ++dd)
#pragma unroll
            for (int r = 0; r < 8; ++r) o[dd][r] *= al[r];

        // ---- transpose P (C-layout -> A-fragment) via wave-private LDS.
        // DS ops from the same wave are executed in order (CDNA5 DScnt),
        // so no block barrier is needed for this private buffer.
#pragma unroll
        for (int r = 0; r < 8; ++r) {
            int row = r + (half << 3);
            pb[row * 33 + mr]      = s0[r];
            pb[row * 33 + 16 + mr] = s1[r];
        }
        v16h pf;
        {
            const float* rp = pb + mr * 33 + (half << 3);
#pragma unroll
            for (int i = 0; i < 8; ++i) {
                pf[i]     = (_Float16)rp[i];
                pf[i + 8] = (_Float16)rp[i + 16];
            }
        }

        // ---- O += P (16x32) * V (32x128); vT layout makes B-frags contiguous
        const _Float16* vbase = vT + (size_t)(b * NKV + kh) * DD * SS;
#pragma unroll
        for (int dd = 0; dd < 8; ++dd) {
            const _Float16* vcol = vbase + (size_t)(dd * 16 + mr) * SS;
            v16h vf = b_frag(vcol, j0, lane);
            o[dd] = wmma16(pf, vf, o[dd]);
        }
    }

    // ---- epilogue: O / l -> ctx[b, s, h*D + d] (f16)
#pragma unroll
    for (int dd = 0; dd < 8; ++dd)
#pragma unroll
        for (int r = 0; r < 8; ++r) {
            int row = r + (half << 3);
            float val = o[dd][r] / lstat[r];
            ctx[(size_t)(b * SS + q0 + row) * (NH * DD) + h * DD + dd * 16 + mr] =
                (_Float16)val;
        }
}

// ---------------------------------------------------------------------------
// Workspace layout (bytes), with overlays (kernels serialize on the stream):
//   [0)            hidden_f16 (16 MiB)   -> later q_f16   (16 MiB)
//   [16777216)     wqkv_f16   (12 MiB)   -> later k_f16 (4 MiB) + vT (4 MiB)
//   [29360128)     wo_f16     (8 MiB)
//   [37748736)     qkv_f32    (48 MiB)   -> later ctx_f16 (16 MiB)
// peak = 88,080,384 bytes
// ---------------------------------------------------------------------------
extern "C" void kernel_launch(void* const* d_in, const int* in_sizes, int n_in,
                              void* d_out, int out_size, void* d_ws, size_t ws_size,
                              hipStream_t stream) {
    (void)in_sizes; (void)n_in; (void)out_size; (void)ws_size;

    const float* hidden    = (const float*)d_in[0];
    const int*   positions = (const int*)d_in[1];
    const float* wqkv      = (const float*)d_in[2];
    const float* wo        = (const float*)d_in[3];
    const float* qnw       = (const float*)d_in[4];
    const float* knw       = (const float*)d_in[5];
    float*       out       = (float*)d_out;

    char* ws = (char*)d_ws;
    _Float16* hidden16 = (_Float16*)(ws);
    _Float16* wqkv16   = (_Float16*)(ws + 16777216);
    _Float16* wo16     = (_Float16*)(ws + 29360128);
    float*    qkv32    = (float*)   (ws + 37748736);
    // overlays
    _Float16* q16   = (_Float16*)(ws);                     // over hidden16
    _Float16* k16   = (_Float16*)(ws + 16777216);          // over wqkv16
    _Float16* vT16  = (_Float16*)(ws + 16777216 + 4194304);
    _Float16* ctx16 = (_Float16*)(ws + 37748736);          // over qkv32

    // 0) f32 -> f16 conversions
    {
        int n4 = (MTOT * HID) / 4;
        f32_to_f16_kernel<<<(n4 + 255) / 256, 256, 0, stream>>>(hidden, hidden16, n4);
        n4 = (NQKV * HID) / 4;
        f32_to_f16_kernel<<<(n4 + 255) / 256, 256, 0, stream>>>(wqkv, wqkv16, n4);
        n4 = (HID * HID) / 4;
        f32_to_f16_kernel<<<(n4 + 255) / 256, 256, 0, stream>>>(wo, wo16, n4);
    }

    // 1) QKV projection: qkv[4096,3072] = hidden[4096,2048] * wqkv[3072,2048]^T
    gemm_f16_nt<<<dim3(NQKV / 256, MTOT / 64), 256, 0, stream>>>(
        hidden16, wqkv16, qkv32, MTOT, NQKV, HID);

    // 2) RMSNorm + RoPE + pack (one wave per token-head; 4096*24/8 blocks)
    pack_qkv_kernel<<<(MTOT * 24) / 8, 256, 0, stream>>>(
        qkv32, positions, qnw, knw, q16, k16, vT16);

    // 3) attention
    attn_kernel<<<dim3(SS / 128, NH, BB), 256, 0, stream>>>(q16, k16, vT16, ctx16);

    // 4) output projection: out[4096,2048] = ctx[4096,2048] * wo[2048,2048]^T
    gemm_f16_nt<<<dim3(HID / 256, MTOT / 64), 256, 0, stream>>>(
        ctx16, wo16, out, MTOT, HID, HID);
}
